// FBAMSoftMemoryAgent_14723147891398
// MI455X (gfx1250) — matrix operand order repaired and checked by reading, not compiled, not run on hardware
//
#include <hip/hip_runtime.h>

// ---- Problem constants (from reference) ----
#define T_STEPS 8192
#define HDIM    512
#define DDIM    128
#define SSLOTS  128
#define AEMB    32
#define FRAME   512
#define CONCAT  672            // FRAME + DDIM + AEMB
#define KTOT    1184           // CONCAT + HDIM
#define KT      37             // KTOT / 32
#define G4H     2048           // 4*HDIM
#define NACT    12
#define NGATE_BLK 16
#define NBLOCKS   17           // 16 gate blocks + 1 specialist

// ---- Workspace layout (bytes) ----
#define WS_SYNC   0            // 2 x uint
#define WS_X      128          // f16[KTOT]
#define WS_GATES  2560         // f32[G4H]
#define WS_PACKW  10752        // f16[128*37*512]
#define PACKW_HALVES (128*37*512)   // 2,424,832
#define WS_WQT    4860416      // f32[512*128]  (Wq transposed)
#define WS_WWT    5122560      // f32[512*128]  (Ww transposed)
#define WS_WCLST  5384704      // f32[512*12]   (W_cls transposed)

// Gate-block LDS: weights slice (8 ntiles * 37 kt * 32 lanes * 16 halves * 2B) + x
#define LDS_W_BYTES 303104
#define SMEM_BYTES  (LDS_W_BYTES + KTOT*2)   // 305,472 < 320KB

typedef __attribute__((ext_vector_type(16))) _Float16 v16h;
typedef __attribute__((ext_vector_type(8)))  _Float16 v8h;
typedef __attribute__((ext_vector_type(8)))  float    v8f;
typedef __attribute__((ext_vector_type(4)))  unsigned u32x4;
typedef __attribute__((ext_vector_type(8)))  int      i32x8;
typedef __attribute__((ext_vector_type(4)))  int      i32x4;

__device__ __forceinline__ float sigmoidf_(float x) {
    return 1.0f / (1.0f + __expf(-x));
}

// --- Tensor Data Mover: 1-D async copy global -> LDS (per CDNA5 D# layout).
// data_size = 8B elements; nbytes must be a multiple of 8 and < 512KB.
__device__ __forceinline__ void tdm_load_1d(const void* gsrc, void* ldst,
                                            unsigned nbytes) {
    unsigned n8 = nbytes >> 3;
    unsigned lds_off = (unsigned)(size_t)ldst;   // low 32 bits = LDS offset
    unsigned long long ga = (unsigned long long)(size_t)gsrc;
    u32x4 g0 = {0u, 0u, 0u, 0u};
    g0[0] = 1u;                                   // count = 1 (valid D#)
    g0[1] = lds_off;                              // lds_addr
    g0[2] = (unsigned)ga;                         // global_addr[31:0]
    g0[3] = (unsigned)((ga >> 32) & 0x01ffffffu) | (2u << 30); // addr[56:32], type=2
    i32x8 g1 = {0, 0, 0, 0, 0, 0, 0, 0};
    g1[0] = 0x00030000;                           // data_size = 3 (8 bytes)
    g1[1] = (int)((n8 & 0xffffu) << 16);          // tensor_dim0[15:0]
    g1[2] = (int)((n8 >> 16) & 0xffffu) | (1 << 16); // dim0[31:16], tensor_dim1=1
    g1[3] = (int)((n8 & 0xffffu) << 16);          // tile_dim0
    g1[4] = 1;                                    // tile_dim1 = 1
    g1[5] = (int)n8;                              // tensor_dim0_stride[31:0]
    i32x4 gz = {0, 0, 0, 0};
#if __has_include(<hip/amd_detail/amd_gfx1250_TDM.h>)
    i32x8 gz8 = {0, 0, 0, 0, 0, 0, 0, 0};
    __builtin_amdgcn_tensor_load_to_lds(g0, g1, gz, gz, gz8, 0);
#else
    __builtin_amdgcn_tensor_load_to_lds(g0, g1, gz, gz, 0);
#endif
}

// Device-scope sense-reversing barrier across NBLOCKS persistent blocks.
__device__ __forceinline__ void gridBarrier(unsigned* cnt, unsigned* gen, unsigned nb) {
    __threadfence();                 // release: flush this thread's stores
    __syncthreads();
    if (threadIdx.x == 0) {
        unsigned g = __hip_atomic_load(gen, __ATOMIC_RELAXED, __HIP_MEMORY_SCOPE_AGENT);
        unsigned old = __hip_atomic_fetch_add(cnt, 1u, __ATOMIC_ACQ_REL, __HIP_MEMORY_SCOPE_AGENT);
        if (old == nb - 1u) {
            __hip_atomic_store(cnt, 0u, __ATOMIC_RELAXED, __HIP_MEMORY_SCOPE_AGENT);
            __hip_atomic_fetch_add(gen, 1u, __ATOMIC_RELEASE, __HIP_MEMORY_SCOPE_AGENT);
        } else {
            while (__hip_atomic_load(gen, __ATOMIC_ACQUIRE, __HIP_MEMORY_SCOPE_AGENT) == g) {
                __builtin_amdgcn_s_sleep(2);
            }
        }
    }
    __syncthreads();
    __threadfence();                 // acquire: invalidate before reading peers' data
}

// Zero the barrier state each launch (ws is poisoned once, never re-poisoned).
__global__ void fbam_init(unsigned char* ws) {
    if (threadIdx.x < 2) ((unsigned*)(ws + WS_SYNC))[threadIdx.x] = 0u;
}

// Pack [W_ih | W_hh] (f32, row-major [2048 x 1184]) into a WMMA f16
// B-fragment layout (contiguous k order, matching A): packed[((nt*37+kt)*32+lane)*16+e] = W[nt*16+lane%16,
// kt*32 + (lane/16)*16 + e], so each lane's 32B B fragment is contiguous.
__global__ void fbam_pack(const float* __restrict__ W_ih,
                          const float* __restrict__ W_hh,
                          unsigned char* ws) {
    _Float16* packed = (_Float16*)(ws + WS_PACKW);
    int idx = blockIdx.x * blockDim.x + threadIdx.x;
    if (idx >= PACKW_HALVES) return;
    int e    = idx & 15;
    int lane = (idx >> 4) & 31;
    int tile = idx >> 9;
    int kt   = tile % KT;
    int nt   = tile / KT;
    int n = nt * 16 + (lane & 15);
    int k = kt * 32 + ((lane >> 4) << 4) + e;
    float v = (k < CONCAT) ? W_ih[(size_t)n * CONCAT + k]
                           : W_hh[(size_t)n * HDIM + (k - CONCAT)];
    packed[idx] = (_Float16)v;
}

// Transpose Wq/Ww/W_cls so the specialist's matvec loads are lane-coalesced.
__global__ void fbam_pack2(const float* __restrict__ Wq,
                           const float* __restrict__ Ww,
                           const float* __restrict__ Wcls,
                           unsigned char* ws) {
    float* wqt = (float*)(ws + WS_WQT);
    float* wwt = (float*)(ws + WS_WWT);
    float* wct = (float*)(ws + WS_WCLST);
    int idx = blockIdx.x * blockDim.x + threadIdx.x;
    if (idx < HDIM * DDIM) {
        int k = idx >> 7, d = idx & 127;
        wqt[k * DDIM + d] = Wq[d * HDIM + k];
        wwt[k * DDIM + d] = Ww[d * HDIM + k];
    }
    if (idx < HDIM * NACT) {
        int k = idx / NACT, j = idx % NACT;
        wct[idx] = Wcls[j * HDIM + k];
    }
}

__global__ void __launch_bounds__(256)
fbam_main(const float* __restrict__ frame_embeds,   // [T,512]
          const int*   __restrict__ actions,        // [T]
          const float* __restrict__ aet,            // [13,32]
          const float* __restrict__ b_ih,           // [2048]
          const float* __restrict__ b_hh,           // [2048]
          const float* __restrict__ b_cls,          // [12]
          const float* __restrict__ memory_init,    // [128,128]
          const float* __restrict__ bq,
          const float* __restrict__ Wk, const float* __restrict__ bk,
          const float* __restrict__ Wv, const float* __restrict__ bv,
          const float* __restrict__ bw,
          const float* __restrict__ wgv, const float* __restrict__ bg,
          unsigned char* ws,
          float* __restrict__ out)                  // [T,12]
{
    extern __shared__ char smem[];
    const int b   = blockIdx.x;
    const int tid = threadIdx.x;

    unsigned* cnt = (unsigned*)(ws + WS_SYNC);
    unsigned* gen = cnt + 1;
    _Float16* x_g     = (_Float16*)(ws + WS_X);
    float*    gates_g = (float*)(ws + WS_GATES);
    const _Float16* packedW = (const _Float16*)(ws + WS_PACKW);
    const float* wqt = (const float*)(ws + WS_WQT);
    const float* wwt = (const float*)(ws + WS_WWT);
    const float* wct = (const float*)(ws + WS_WCLST);

    if (b < NGATE_BLK) {
        // ---------------- Gate blocks: LDS-resident weight slice + WMMA ----
        _Float16* wt = (_Float16*)smem;                  // 8*37*32*16 halves
        _Float16* xl = (_Float16*)(smem + LDS_W_BYTES);  // x in f16, KTOT

        const int lane = tid & 31;
        const int wid  = tid >> 5;          // 8 waves -> 8 N-tiles of 16 outputs
        const int g16  = (lane & 16) ? 8 : 0;

        // One-time TDM DMA of this block's 303KB weight slice into LDS.
        if (wid == 0) {
            tdm_load_1d(packedW + (size_t)b * (8 * KT * 512), wt, LDS_W_BYTES);
            __builtin_amdgcn_s_wait_tensorcnt(0);
        }
        __syncthreads();

        auto mm = [&](int kt, v8f acc) -> v8f {
            // A fragment: x replicated across all 16 rows (batch-1 matvec).
            v8h lo = *(const v8h*)(xl + kt * 32 + 2 * g16);
            v8h hi = *(const v8h*)(xl + kt * 32 + 2 * g16 + 8);
            v16h a = __builtin_shufflevector(lo, hi,
                     0,1,2,3,4,5,6,7,8,9,10,11,12,13,14,15);
            // B fragment: contiguous 32B per lane from LDS.
            v16h bf = *(const v16h*)(wt + ((((wid * KT) + kt) * 32 + lane) << 4));
            return __builtin_amdgcn_wmma_f32_16x16x32_f16(
                       false, a, false, bf, (short)0, acc, false, false);
        };

        for (int t = 0; t < T_STEPS; ++t) {
            gridBarrier(cnt, gen, NBLOCKS);             // A: x ready
            // Stage x (f16, 2368B) into LDS via the Tensor Data Mover.
            if (wid == 0) {
                tdm_load_1d(x_g, xl, KTOT * 2);
                __builtin_amdgcn_s_wait_tensorcnt(0);
            }
            __syncthreads();

            // 4 independent accumulator chains to break WMMA->WMMA RAW.
            v8f s0 = {}, s1 = {}, s2 = {}, s3 = {};
            #pragma unroll 1
            for (int kt = 0; kt < KT - 1; kt += 4) {
                s0 = mm(kt + 0, s0);
                s1 = mm(kt + 1, s1);
                s2 = mm(kt + 2, s2);
                s3 = mm(kt + 3, s3);
            }
            s0 = mm(KT - 1, s0);
            v8f acc = (s0 + s1) + (s2 + s3);

            // Row 0 of C holds the 16 outputs (lanes 0..15, N = lane).
            if (lane < 16)
                gates_g[b * 128 + wid * 16 + lane] = acc[0];
            gridBarrier(cnt, gen, NBLOCKS);             // B: gates ready
        }
    } else {
        // ---------------- Specialist block: attention + LSTM + memory -----
        float* keys = (float*)smem;                    // 128*128
        float* vals = keys + SSLOTS * DDIM;            // 128*128
        float* h    = vals + SSLOTS * DDIM;            // 512
        float* c    = h + HDIM;                        // 512
        float* attn = c + HDIM;                        // 128
        float* qv   = attn + SSLOTS;                   // 128
        float* red  = qv + DDIM;                       // 256
        float* wvv  = red + 256;                       // 128

        for (int i = tid; i < HDIM; i += 256) { h[i] = 0.f; c[i] = 0.f; }
        // keys0 = memory_init @ Wk^T + bk ; vals0 = memory_init @ Wv^T + bv
        for (int i = tid; i < SSLOTS * DDIM; i += 256) {
            int s = i >> 7, d = i & 127;
            float aK = bk[d], aV = bv[d];
            for (int m = 0; m < DDIM; ++m) {
                float mi = memory_init[s * DDIM + m];
                aK += mi * Wk[d * DDIM + m];
                aV += mi * Wv[d * DDIM + m];
            }
            keys[i] = aK; vals[i] = aV;
        }
        __syncthreads();

        const float scale = 0.08838834764831845f;      // 1/sqrt(128)

        for (int t = 0; t < T_STEPS; ++t) {
            const float mask = (t == 0) ? 0.f : 1.f;
            if (tid == 0 && t + 1 < T_STEPS)
                __builtin_prefetch(frame_embeds + (size_t)(t + 1) * FRAME, 0, 1);
            // ---- Phase 1: attention read + build x ----
            if (tid < DDIM) {
                float a = bq[tid];
                for (int k = 0; k < HDIM; ++k) a += h[k] * wqt[k * DDIM + tid];
                qv[tid] = a;
            }
            __syncthreads();
            if (tid < SSLOTS) {
                float a = 0.f;
                for (int d = 0; d < DDIM; ++d) a += keys[tid * DDIM + d] * qv[d];
                attn[tid] = a * scale;
            }
            __syncthreads();
            // softmax over 128
            red[tid] = (tid < SSLOTS) ? attn[tid] : -1e30f;
            __syncthreads();
            for (int off = 128; off >= 1; off >>= 1) {
                if (tid < off) red[tid] = fmaxf(red[tid], red[tid + off]);
                __syncthreads();
            }
            float mx = red[0];
            __syncthreads();
            if (tid < SSLOTS) attn[tid] = __expf(attn[tid] - mx);
            __syncthreads();
            red[tid] = (tid < SSLOTS) ? attn[tid] : 0.f;
            __syncthreads();
            for (int off = 128; off >= 1; off >>= 1) {
                if (tid < off) red[tid] += red[tid + off];
                __syncthreads();
            }
            float inv = 1.f / red[0];
            __syncthreads();
            if (tid < SSLOTS) attn[tid] *= inv;
            __syncthreads();

            // x = [frame | mask*retrieved | prev_emb | h] as f16 in global
            const float* fe = frame_embeds + (size_t)t * FRAME;
            for (int i = tid; i < FRAME; i += 256) x_g[i] = (_Float16)fe[i];
            if (tid < DDIM) {
                float r = 0.f;
                for (int s = 0; s < SSLOTS; ++s) r += attn[s] * vals[s * DDIM + tid];
                x_g[FRAME + tid] = (_Float16)(mask * r);
            }
            {
                int pa = (t == 0) ? NACT : actions[t - 1];
                if (tid < AEMB) x_g[FRAME + DDIM + tid] = (_Float16)aet[pa * AEMB + tid];
            }
            for (int i = tid; i < HDIM; i += 256) x_g[CONCAT + i] = (_Float16)h[i];

            gridBarrier(cnt, gen, NBLOCKS);             // A: x published
            gridBarrier(cnt, gen, NBLOCKS);             // B: gates ready

            // ---- Phase 3: LSTM pointwise, write, classify ----
            for (int i = tid; i < HDIM; i += 256) {
                float gi = gates_g[i]            + b_ih[i]            + b_hh[i];
                float gf = gates_g[HDIM + i]     + b_ih[HDIM + i]     + b_hh[HDIM + i];
                float gg = gates_g[2 * HDIM + i] + b_ih[2 * HDIM + i] + b_hh[2 * HDIM + i];
                float go = gates_g[3 * HDIM + i] + b_ih[3 * HDIM + i] + b_hh[3 * HDIM + i];
                float cn = sigmoidf_(gf) * c[i] + sigmoidf_(gi) * tanhf(gg);
                c[i] = cn;
                h[i] = sigmoidf_(go) * tanhf(cn);
            }
            __syncthreads();
            // write gate scalar = sigmoid(h_new . wg + bg)
            {
                float p = 0.f;
                for (int i = tid; i < HDIM; i += 256) p += h[i] * wgv[i];
                red[tid] = p;
            }
            __syncthreads();
            for (int off = 128; off >= 1; off >>= 1) {
                if (tid < off) red[tid] += red[tid + off];
                __syncthreads();
            }
            float wgate = sigmoidf_(red[0] + bg[0]);
            __syncthreads();
            // write_vec and logits in parallel thread groups (coalesced reads)
            if (tid < DDIM) {
                float a = bw[tid];
                for (int k = 0; k < HDIM; ++k) a += h[k] * wwt[k * DDIM + tid];
                wvv[tid] = a;
            } else if (tid < DDIM + NACT) {
                int j = tid - DDIM;
                float a = b_cls[j];
                for (int k = 0; k < HDIM; ++k) a += h[k] * wct[k * NACT + j];
                out[(size_t)t * NACT + j] = a;
            }
            __syncthreads();
            // gated memory write into keys/vals (LDS-resident state)
            for (int i = tid; i < SSLOTS * DDIM; i += 256) {
                int s = i >> 7, d = i & 127;
                float w = mask * wgate * attn[s];
                float wv = wvv[d];
                vals[i] = vals[i] * (1.f - w) + w * wv;
                keys[i] = keys[i] * (1.f - w) + w * wv;
            }
            __syncthreads();
        }
    }
}

extern "C" void kernel_launch(void* const* d_in, const int* in_sizes, int n_in,
                              void* d_out, int out_size, void* d_ws, size_t ws_size,
                              hipStream_t stream) {
    const float* frame_embeds = (const float*)d_in[0];
    const int*   actions      = (const int*)d_in[1];
    const float* aet          = (const float*)d_in[2];
    const float* W_ih         = (const float*)d_in[3];
    const float* W_hh         = (const float*)d_in[4];
    const float* b_ih         = (const float*)d_in[5];
    const float* b_hh         = (const float*)d_in[6];
    const float* W_cls        = (const float*)d_in[7];
    const float* b_cls        = (const float*)d_in[8];
    const float* memory_init  = (const float*)d_in[9];
    const float* Wq = (const float*)d_in[10];
    const float* bq = (const float*)d_in[11];
    const float* Wk = (const float*)d_in[12];
    const float* bk = (const float*)d_in[13];
    const float* Wv = (const float*)d_in[14];
    const float* bv = (const float*)d_in[15];
    const float* Ww = (const float*)d_in[16];
    const float* bw = (const float*)d_in[17];
    const float* wgv = (const float*)d_in[18];
    const float* bg  = (const float*)d_in[19];
    unsigned char* ws = (unsigned char*)d_ws;
    float* out = (float*)d_out;

    fbam_init<<<1, 64, 0, stream>>>(ws);
    fbam_pack<<<(PACKW_HALVES + 255) / 256, 256, 0, stream>>>(W_ih, W_hh, ws);
    fbam_pack2<<<(HDIM * DDIM + 255) / 256, 256, 0, stream>>>(Wq, Ww, W_cls, ws);
    fbam_main<<<NBLOCKS, 256, SMEM_BYTES, stream>>>(
        frame_embeds, actions, aet, b_ih, b_hh, b_cls, memory_init,
        bq, Wk, bk, Wv, bv, bw, wgv, bg, ws, out);
}
